// HabanaAttentionImpl_7937099563214
// MI455X (gfx1250) — matrix operand order repaired and checked by reading, not compiled, not run on hardware
//
#include <hip/hip_runtime.h>

#define NUM_HEADS    32
#define HEAD_SIZE    128
#define NUM_KV_HEADS 8
#define GQA          4          // NUM_HEADS / NUM_KV_HEADS
#define ATT_SCALE    0.08838834764831845f
#define LOG2E        1.4426950408889634f
#define BATCH        4
#define SEQ          1024
#define NUM_BLOCKS   64
#define BLOCK_SZ     128

typedef __attribute__((ext_vector_type(16))) _Float16 v16h;
typedef __attribute__((ext_vector_type(8)))  _Float16 v8h;
typedef __attribute__((ext_vector_type(4)))  _Float16 v4h;
typedef __attribute__((ext_vector_type(8)))  float    v8f;

union FragA { v16h v; v8h h[2]; };

// ---------------------------------------------------------------------------
// Kernel 1: pass-through copy of the input caches (float4-wide, b128 traffic).
// ---------------------------------------------------------------------------
__global__ void cache_copy_kernel(const float4* __restrict__ kc_in,
                                  const float4* __restrict__ vc_in,
                                  float4* __restrict__ kc_out,
                                  float4* __restrict__ vc_out, int n4) {
    int i = blockIdx.x * blockDim.x + threadIdx.x;
    if (i < n4) {
        kc_out[i] = kc_in[i];
        vc_out[i] = vc_in[i];
    }
}

// ---------------------------------------------------------------------------
// Kernel 2: scatter K/V into the paged caches by slot_mapping (float4-wide).
// cache layout [blk][off][hkv][d]: blk*128+off == slot, token row = 1024 f32
// = 256 float4 chunks.
// ---------------------------------------------------------------------------
__global__ void cache_scatter_kernel(const float4* __restrict__ k,
                                     const float4* __restrict__ v,
                                     const int* __restrict__ slot_mapping,
                                     float4* __restrict__ kc_out,
                                     float4* __restrict__ vc_out, int n4) {
    int i = blockIdx.x * blockDim.x + threadIdx.x;
    if (i >= n4) return;
    int tok = i >> 8;             // / 256 chunks per token
    int e4  = i & 255;
    int s   = slot_mapping[tok];
    size_t dst = (size_t)s * 256 + e4;
    kc_out[dst] = k[i];
    vc_out[dst] = v[i];
}

// ---------------------------------------------------------------------------
// Kernel 3: causal GQA flash attention using v_wmma_f32_16x16x32_f16.
// Grid: (S/64, NUM_HEADS, B). 128 threads = 4 wave32; each wave owns 16 q-rows.
// Softmax runs in log2 domain (Q pre-scaled by SCALE*log2e, exp2f everywhere).
// ---------------------------------------------------------------------------
__launch_bounds__(128, 1)
__global__ void gqa_attn_kernel(const float* __restrict__ q,
                                const float* __restrict__ k,
                                const float* __restrict__ v,
                                float* __restrict__ out) {
    __shared__ __align__(16) _Float16 Ks[32][128];      // K tile   [key][d]
    __shared__ __align__(16) _Float16 Vt[128][32];      // V tile^T [d][key]
    __shared__ __align__(16) _Float16 Qs[4][16][128];   // per-wave Q tile (f16, pre-scaled)
    __shared__ __align__(16) _Float16 Ps[4][16][32];    // per-wave P bounce buffer

    const int qb   = blockIdx.x;          // query tile of 64 rows
    const int head = blockIdx.y;          // 0..31
    const int b    = blockIdx.z;
    const int hkv  = head / GQA;

    const int tid  = threadIdx.x;
    const int wave = tid >> 5;
    const int lane = tid & 31;
    const int half = lane >> 4;           // lane half selects row/K sub-range
    const int ln   = lane & 15;           // matrix row (A) or column (B/C/D)

    const int q0w = qb * 64 + wave * 16;  // first query row of this wave
    const float qscale = ATT_SCALE * LOG2E;

    // ---- stage this wave's Q rows (f32 -> f16, b128 loads, scale folded) ----
    for (int idx = lane; idx < 16 * 32; idx += 32) {    // 512 float4 chunks
        int r = idx >> 5, d4 = idx & 31;
        const float4 f = *(const float4*)&q[((size_t)(b * SEQ + q0w + r)) * (NUM_HEADS * HEAD_SIZE)
                                            + head * HEAD_SIZE + d4 * 4];
        v4h h = { (_Float16)(f.x * qscale), (_Float16)(f.y * qscale),
                  (_Float16)(f.z * qscale), (_Float16)(f.w * qscale) };
        *(v4h*)&Qs[wave][r][d4 * 4] = h;
    }
    __syncthreads();

    // A-fragments of Q: element i -> K = (i/8)*16 + half*8 + i%8  (ISA 7.12.2)
    FragA qa[4];
#pragma unroll
    for (int c = 0; c < 4; ++c) {
        qa[c].h[0] = *(const v8h*)&Qs[wave][ln][c * 32 + half * 8];
        qa[c].h[1] = *(const v8h*)&Qs[wave][ln][c * 32 + 16 + half * 8];
    }

    const v8f zero8 = {0.f, 0.f, 0.f, 0.f, 0.f, 0.f, 0.f, 0.f};
    v8f   O[8];
    float m[8], l[8];
#pragma unroll
    for (int c = 0; c < 8; ++c) O[c] = zero8;
#pragma unroll
    for (int j = 0; j < 8; ++j) { m[j] = -1e30f; l[j] = 0.f; }

    const int nkb = 2 * qb + 2;           // causal: blocks of 32 keys
    for (int kbi = 0; kbi < nkb; ++kbi) {
        __syncthreads();                  // protect LDS tiles of prev iter
        // ---- cooperatively stage 32 keys of K and V^T (b128 loads, f16) ----
        for (int idx = tid; idx < 32 * 32; idx += 128) { // 1024 float4 chunks
            int key = idx >> 5, d4 = idx & 31;
            size_t gi = ((size_t)(b * SEQ + kbi * 32 + key)) * (NUM_KV_HEADS * HEAD_SIZE)
                        + hkv * HEAD_SIZE + d4 * 4;
            const float4 kf = *(const float4*)&k[gi];
            const float4 vf = *(const float4*)&v[gi];
            v4h kh = { (_Float16)kf.x, (_Float16)kf.y, (_Float16)kf.z, (_Float16)kf.w };
            *(v4h*)&Ks[key][d4 * 4] = kh;
            int d = d4 * 4;
            Vt[d + 0][key] = (_Float16)vf.x;
            Vt[d + 1][key] = (_Float16)vf.y;
            Vt[d + 2][key] = (_Float16)vf.z;
            Vt[d + 3][key] = (_Float16)vf.w;
            if (kbi + 1 < nkb) {          // prefetch next key block (global_prefetch_b8)
                __builtin_prefetch(&k[gi + 32 * (NUM_KV_HEADS * HEAD_SIZE)], 0, 0);
                __builtin_prefetch(&v[gi + 32 * (NUM_KV_HEADS * HEAD_SIZE)], 0, 0);
            }
        }
        __syncthreads();

        // ---- S = Q * K^T for two 16-key column tiles ----
        v8f s0 = zero8, s1 = zero8;
#pragma unroll
        for (int c = 0; c < 4; ++c) {
            FragA kb0, kb1;               // B-fragment: lane=key col, elem i -> d = (i/8)*16 + half*8 + i%8
            kb0.h[0] = *(const v8h*)&Ks[ln][c * 32 + half * 8];
            kb0.h[1] = *(const v8h*)&Ks[ln][c * 32 + 16 + half * 8];
            kb1.h[0] = *(const v8h*)&Ks[16 + ln][c * 32 + half * 8];
            kb1.h[1] = *(const v8h*)&Ks[16 + ln][c * 32 + 16 + half * 8];
            s0 = __builtin_amdgcn_wmma_f32_16x16x32_f16(false, qa[c].v, false, kb0.v,
                                                        (short)0, s0, false, false);
            s1 = __builtin_amdgcn_wmma_f32_16x16x32_f16(false, qa[c].v, false, kb1.v,
                                                        (short)0, s1, false, false);
        }

        // ---- causal mask + online softmax (log2 domain) on the C-fragment ----
        const int kbase = kbi * 32;
#pragma unroll
        for (int j = 0; j < 8; ++j) {
            int qg = q0w + half * 8 + j;  // this element's query row
            if (kbase + ln > qg)      s0[j] = -1e30f;
            if (kbase + 16 + ln > qg) s1[j] = -1e30f;

            float rmax = fmaxf(s0[j], s1[j]);   // reduce over 16 key columns (lanes)
            rmax = fmaxf(rmax, __shfl_xor(rmax, 1, 32));
            rmax = fmaxf(rmax, __shfl_xor(rmax, 2, 32));
            rmax = fmaxf(rmax, __shfl_xor(rmax, 4, 32));
            rmax = fmaxf(rmax, __shfl_xor(rmax, 8, 32));

            float mn    = fmaxf(m[j], rmax);
            float alpha = exp2f(m[j] - mn);     // v_exp_f32, no ln2 fixup
            m[j] = mn;

            float p0 = exp2f(s0[j] - mn);
            float p1 = exp2f(s1[j] - mn);

            float rsum = p0 + p1;
            rsum += __shfl_xor(rsum, 1, 32);
            rsum += __shfl_xor(rsum, 2, 32);
            rsum += __shfl_xor(rsum, 4, 32);
            rsum += __shfl_xor(rsum, 8, 32);
            l[j] = l[j] * alpha + rsum;

            // bounce P through LDS to re-layout C-frag -> A-frag
            Ps[wave][half * 8 + j][ln]      = (_Float16)p0;
            Ps[wave][half * 8 + j][16 + ln] = (_Float16)p1;

#pragma unroll
            for (int c = 0; c < 8; ++c) O[c][j] *= alpha;   // rescale accumulator
        }

        // ---- O += P * V ----
        FragA pa;
        pa.h[0] = *(const v8h*)&Ps[wave][ln][half * 8];
        pa.h[1] = *(const v8h*)&Ps[wave][ln][16 + half * 8];
#pragma unroll
        for (int c = 0; c < 8; ++c) {
            FragA vb;                     // B-fragment: lane = d col, elem i -> key (i/8)*16 + half*8 + i%8
            vb.h[0] = *(const v8h*)&Vt[c * 16 + ln][half * 8];
            vb.h[1] = *(const v8h*)&Vt[c * 16 + ln][16 + half * 8];
            O[c] = __builtin_amdgcn_wmma_f32_16x16x32_f16(false, pa.v, false, vb.v,
                                                          (short)0, O[c], false, false);
        }
    }

    // ---- epilogue: divide by l, store f32 ----
#pragma unroll
    for (int j = 0; j < 8; ++j) {
        float inv = 1.0f / l[j];
        int qg = q0w + half * 8 + j;
        size_t base = ((size_t)(b * SEQ + qg)) * (NUM_HEADS * HEAD_SIZE) + head * HEAD_SIZE;
#pragma unroll
        for (int c = 0; c < 8; ++c)
            out[base + c * 16 + ln] = O[c][j] * inv;
    }
}

// ---------------------------------------------------------------------------
extern "C" void kernel_launch(void* const* d_in, const int* in_sizes, int n_in,
                              void* d_out, int out_size, void* d_ws, size_t ws_size,
                              hipStream_t stream) {
    const float* q    = (const float*)d_in[0];
    const float* k    = (const float*)d_in[1];
    const float* v    = (const float*)d_in[2];
    const float* kc   = (const float*)d_in[3];
    const float* vc   = (const float*)d_in[4];
    const int*   slot = (const int*)d_in[5];

    float* out = (float*)d_out;
    const size_t attn_n  = (size_t)BATCH * SEQ * NUM_HEADS * HEAD_SIZE;              // 16.78M
    const size_t cache_n = (size_t)NUM_BLOCKS * BLOCK_SZ * NUM_KV_HEADS * HEAD_SIZE; // 8.39M
    float* out_kc = out + attn_n;
    float* out_vc = out_kc + cache_n;

    // 1) cache pass-through copy (b128)
    {
        int n4 = (int)(cache_n / 4);
        cache_copy_kernel<<<(n4 + 255) / 256, 256, 0, stream>>>(
            (const float4*)kc, (const float4*)vc, (float4*)out_kc, (float4*)out_vc, n4);
    }
    // 2) scatter new K/V by slot_mapping (b128)
    {
        int n4 = BATCH * SEQ * NUM_KV_HEADS * HEAD_SIZE / 4;
        cache_scatter_kernel<<<(n4 + 255) / 256, 256, 0, stream>>>(
            (const float4*)k, (const float4*)v, slot, (float4*)out_kc, (float4*)out_vc, n4);
    }
    // 3) causal GQA flash attention (WMMA f16 -> f32)
    {
        dim3 grid(SEQ / 64, NUM_HEADS, BATCH);
        gqa_attn_kernel<<<grid, 128, 0, stream>>>(q, k, v, out);
    }
}
